// PatchLoss_1030792151609
// MI455X (gfx1250) — compile-verified, matrix operand order checked by reading, not run on hardware
//
#include <hip/hip_runtime.h>
#include <hip/hip_bf16.h>

// ---------------------------------------------------------------------------
// Problem constants (from the reference)
// ---------------------------------------------------------------------------
#define NB     2            // batch
#define CCH    128          // channels
#define HIN    64
#define WIN    64
#define H2     128          // shape/color feat H
#define W2     128
#define HP     62           // 64-3+1
#define WP     62
#define NQ     (HP*WP)      // 3844 queries per batch
#define HS     126          // 128-3+1
#define WS     126
#define NU     (HS*WS)      // 15876 candidates per batch
#define KDIM   (CCH*9)      // 1152

// ---------------------------------------------------------------------------
// GEMM tiling: block = 8 waves (256 thr), tile 64M x 256U, K slab = 64,
// double-buffered LDS. Per wave: 32M x 64U (2 m-frags x 4 u-frags).
// ---------------------------------------------------------------------------
#define M_BLK  64
#define U_BLK  256
#define MBLKS  61                       // ceil(3844/64)
#define MPAD   (MBLKS*M_BLK)            // 3904
#define UBLKS  63                       // ceil(15876/256)
#define UPAD   (UBLKS*U_BLK)            // 16128
#define KSLAB  64
#define NSLAB  (KDIM/KSLAB)             // 18
#define LDS_STRIDE 72                   // 64 + 8 pad (elements) -> conflict-free b128
#define A_ELEMS (M_BLK*LDS_STRIDE)      // 4608
#define B_ELEMS (U_BLK*LDS_STRIDE)      // 18432
#define BUF_ELEMS (A_ELEMS + B_ELEMS)   // 23040
#define SMEM_BYTES (2*BUF_ELEMS*2)      // 92160 B (dynamic LDS; WGP has 320 KB)

typedef __attribute__((ext_vector_type(16))) __bf16 v16bf;
typedef __attribute__((ext_vector_type(8)))  float  v8f;

union Frag {
    uint4 u[2];
    v16bf v;
};

// Monotone map float -> uint32 so unsigned compare == float compare.
__device__ __forceinline__ unsigned int f2key(float f) {
    unsigned int b = __float_as_uint(f);
    return b ^ ((b & 0x80000000u) ? 0xFFFFFFFFu : 0x80000000u);
}

__device__ __forceinline__ unsigned long long shfl_xor_u64(unsigned long long x, int m) {
    unsigned int lo = (unsigned int)(x & 0xFFFFFFFFull);
    unsigned int hi = (unsigned int)(x >> 32);
    lo = __shfl_xor(lo, m, 32);
    hi = __shfl_xor(hi, m, 32);
    return (((unsigned long long)hi) << 32) | lo;
}

// ---------------------------------------------------------------------------
// Init: zero argmax table and the scalar output
// ---------------------------------------------------------------------------
__global__ void init_kernel(unsigned long long* __restrict__ amax, float* __restrict__ out) {
    int i = blockIdx.x * blockDim.x + threadIdx.x;
    if (i == 0) out[0] = 0.0f;
    if (i < NB * NQ) amax[i] = 0ull;
}

// ---------------------------------------------------------------------------
// Pack ref patches -> Q (MPAD x KDIM, row-major bf16). k = c*9 + a*3 + b
// ---------------------------------------------------------------------------
__global__ void pack_q_kernel(const float* __restrict__ ref,
                              __hip_bfloat16* __restrict__ Q, int n) {
    int idx = blockIdx.x * blockDim.x + threadIdx.x;
    if (idx >= MPAD * KDIM) return;
    int q = idx / KDIM;
    int k = idx % KDIM;
    float v = 0.0f;
    if (q < NQ) {
        int i = q / WP, j = q % WP;
        int c = k / 9, r = k % 9, a = r / 3, b = r % 3;
        v = ref[(((long)n * CCH + c) * HIN + (i + a)) * WIN + (j + b)];
    }
    Q[idx] = __float2bfloat16(v);
}

// ---------------------------------------------------------------------------
// Pack shape patches -> S (UPAD x KDIM, row-major bf16). Same k ordering.
// ---------------------------------------------------------------------------
__global__ void pack_s_kernel(const float* __restrict__ shp,
                              __hip_bfloat16* __restrict__ S, int n) {
    int idx = blockIdx.x * blockDim.x + threadIdx.x;
    if (idx >= UPAD * KDIM) return;
    int u = idx / KDIM;
    int k = idx % KDIM;
    float v = 0.0f;
    if (u < NU) {
        int p = u / WS, qq = u % WS;
        int c = k / 9, r = k % 9, a = r / 3, b = r % 3;
        v = shp[(((long)n * CCH + c) * H2 + (p + a)) * W2 + (qq + b)];
    }
    S[idx] = __float2bfloat16(v);
}

// ---------------------------------------------------------------------------
// Stage one K slab (64 wide) of the A (64 rows) and B (256 rows) tiles into
// LDS. 256 threads: 2 b128 chunks of A + 8 b128 chunks of B each.
// ---------------------------------------------------------------------------
__device__ __forceinline__ void stage_slab(const __hip_bfloat16* __restrict__ Q,
                                           const __hip_bfloat16* __restrict__ S,
                                           __hip_bfloat16* __restrict__ buf,
                                           int m0, int u0, int ks, int tid) {
    uint4 ta[2], tb[8];
#pragma unroll
    for (int i = 0; i < 2; ++i) {
        int id = tid + i * 256;
        int row = id >> 3, c8 = id & 7;          // 8 chunks per 64-wide row
        ta[i] = *(const uint4*)(Q + (long)(m0 + row) * KDIM + ks + c8 * 8);
    }
#pragma unroll
    for (int i = 0; i < 8; ++i) {
        int id = tid + i * 256;
        int row = id >> 3, c8 = id & 7;
        tb[i] = *(const uint4*)(S + (long)(u0 + row) * KDIM + ks + c8 * 8);
    }
    if (ks + KSLAB < KDIM) {   // speculative next-slab prefetch (gfx1250 path)
        __builtin_prefetch(Q + (long)(m0 + (tid >> 3)) * KDIM + ks + KSLAB + (tid & 7) * 8, 0, 1);
        __builtin_prefetch(S + (long)(u0 + (tid >> 3)) * KDIM + ks + KSLAB + (tid & 7) * 8, 0, 1);
    }
#pragma unroll
    for (int i = 0; i < 2; ++i) {
        int id = tid + i * 256;
        int row = id >> 3, c8 = id & 7;
        *(uint4*)(buf + row * LDS_STRIDE + c8 * 8) = ta[i];
    }
#pragma unroll
    for (int i = 0; i < 8; ++i) {
        int id = tid + i * 256;
        int row = id >> 3, c8 = id & 7;
        *(uint4*)(buf + A_ELEMS + row * LDS_STRIDE + c8 * 8) = tb[i];
    }
}

// ---------------------------------------------------------------------------
// LDS-tiled GEMM (bf16 WMMA, f32 acc) + fused argmax over candidates.
// A/B fragment per-lane layout (16-bit, 16x32): lane<16  -> row, K {0..7,16..23}
//                                               lane>=16 -> row, K {8..15,24..31}
// -> two b128 loads per fragment (global while packing; ds here).
// ---------------------------------------------------------------------------
__global__ void __launch_bounds__(256) gemm_argmax_kernel(
    const __hip_bfloat16* __restrict__ Q,
    const __hip_bfloat16* __restrict__ S,
    unsigned long long* __restrict__ amax) {

    extern __shared__ __align__(16) __hip_bfloat16 smem[];

    const int tid  = threadIdx.x;
    const int lane = tid & 31;
    const int wave = tid >> 5;
    const int half = lane >> 4;          // 0/1
    const int l16  = lane & 15;
    const int koff = half * 8;           // element offset within 32-wide K step

    const int wm = wave & 1;             // 2 m-groups of waves
    const int wu = wave >> 1;            // 4 u-groups of waves

    const int m0 = blockIdx.x * M_BLK;
    const int u0 = blockIdx.y * U_BLK;

    v8f acc[2][4];
#pragma unroll
    for (int f = 0; f < 2; ++f)
#pragma unroll
        for (int g = 0; g < 4; ++g)
            acc[f][g] = (v8f){};

    __hip_bfloat16* buf0 = smem;
    __hip_bfloat16* buf1 = smem + BUF_ELEMS;

    stage_slab(Q, S, buf0, m0, u0, 0, tid);
    __syncthreads();

    for (int s = 0; s < NSLAB; ++s) {
        __hip_bfloat16* cur = (s & 1) ? buf1 : buf0;
        if (s + 1 < NSLAB)   // fill the other buffer while computing this one
            stage_slab(Q, S, (s & 1) ? buf0 : buf1, m0, u0, (s + 1) * KSLAB, tid);

        __hip_bfloat16* bA = cur;
        __hip_bfloat16* bB = cur + A_ELEMS;

#pragma unroll
        for (int t = 0; t < 2; ++t) {    // two 32-wide K steps per slab
            Frag a0, a1;
            const __hip_bfloat16* pa = bA + (wm * 32 + l16) * LDS_STRIDE + t * 32 + koff;
            a0.u[0] = *(const uint4*)(pa);
            a0.u[1] = *(const uint4*)(pa + 16);
            a1.u[0] = *(const uint4*)(pa + 16 * LDS_STRIDE);
            a1.u[1] = *(const uint4*)(pa + 16 * LDS_STRIDE + 16);
#pragma unroll
            for (int g = 0; g < 4; ++g) {
                Frag b;
                const __hip_bfloat16* pb =
                    bB + (wu * 64 + g * 16 + l16) * LDS_STRIDE + t * 32 + koff;
                b.u[0] = *(const uint4*)(pb);
                b.u[1] = *(const uint4*)(pb + 16);
                acc[0][g] = __builtin_amdgcn_wmma_f32_16x16x32_bf16(
                    false, a0.v, false, b.v, (short)0, acc[0][g], false, false);
                acc[1][g] = __builtin_amdgcn_wmma_f32_16x16x32_bf16(
                    false, a1.v, false, b.v, (short)0, acc[1][g], false, false);
            }
        }
        __syncthreads();
    }

    // D layout: lane holds column n = l16 (u = u0 + wu*64 + g*16 + l16);
    // VGPR v holds row m = v + 8*half (within the 16-row fragment).
    // xor-shuffles with mask<16 stay inside each lane half.
#pragma unroll
    for (int f = 0; f < 2; ++f) {
#pragma unroll
        for (int v = 0; v < 8; ++v) {
            const int m = m0 + wm * 32 + f * 16 + v + 8 * half;
            unsigned long long best = 0ull;
#pragma unroll
            for (int g = 0; g < 4; ++g) {
                int u = u0 + wu * 64 + g * 16 + l16;
                unsigned long long p =
                    (((unsigned long long)f2key(acc[f][g][v])) << 32) | (unsigned)u;
                if (u < NU && p > best) best = p;
            }
#pragma unroll
            for (int sft = 1; sft < 16; sft <<= 1) {
                unsigned long long o = shfl_xor_u64(best, sft);
                if (o > best) best = o;
            }
            if (l16 == 0 && m < NQ && best != 0ull)
                atomicMax(&amax[m], best);
        }
    }
}

// ---------------------------------------------------------------------------
// L1 loss against matched color patch, clip, scalar accumulate.
// One 128-thread block per (n,i,j); thread = channel.
// ---------------------------------------------------------------------------
__global__ void __launch_bounds__(128) loss_kernel(
    const float* __restrict__ outf, const float* __restrict__ colf,
    const unsigned long long* __restrict__ amax, float* __restrict__ out) {

    __shared__ float red[128];
    const int blk = blockIdx.x;          // 0 .. NB*NQ-1
    const int n = blk / NQ;
    const int q = blk % NQ;
    const int i = q / WP, j = q % WP;

    const unsigned long long p = amax[(long)n * NQ + q];
    const unsigned int u = (unsigned int)(p & 0xFFFFFFFFull);
    const int pi = u / WS, pj = u % WS;

    const int c = threadIdx.x;
    const float* o  = outf + (((long)n * CCH + c) * HIN + i) * WIN + j;
    const float* cl = colf + (((long)n * CCH + c) * H2  + pi) * W2  + pj;

    float s = 0.0f;
#pragma unroll
    for (int a = 0; a < 3; ++a)
#pragma unroll
        for (int b = 0; b < 3; ++b)
            s += fabsf(o[a * WIN + b] - cl[a * W2 + b]);

    red[c] = s;
    __syncthreads();
    for (int st = 64; st > 0; st >>= 1) {
        if (c < st) red[c] += red[c + st];
        __syncthreads();
    }
    if (c == 0) {
        float l1 = red[0];
        l1 = fminf(fmaxf(l1, 0.0f), 1000.0f);
        atomicAdd(out, l1);
    }
}

// ---------------------------------------------------------------------------
// Launch
// ---------------------------------------------------------------------------
extern "C" void kernel_launch(void* const* d_in, const int* in_sizes, int n_in,
                              void* d_out, int out_size, void* d_ws, size_t ws_size,
                              hipStream_t stream) {
    const float* output_feat = (const float*)d_in[0];
    const float* ref_feat    = (const float*)d_in[1];
    const float* shape_feat  = (const float*)d_in[2];
    const float* color_feat  = (const float*)d_in[3];
    float* out = (float*)d_out;

    // Scratch layout (batches processed sequentially, reusing Q/S):
    //   Q: MPAD x KDIM bf16  (~9.0 MB)
    //   S: UPAD x KDIM bf16  (~37.2 MB)   -> both resident in the 192 MB L2
    //   amax: NB*NQ u64      (~60 KB)
    char* ws = (char*)d_ws;
    size_t szQ  = (size_t)MPAD * KDIM * sizeof(__hip_bfloat16);
    size_t offS = (szQ + 255) & ~(size_t)255;
    size_t szS  = (size_t)UPAD * KDIM * sizeof(__hip_bfloat16);
    size_t offA = (offS + szS + 255) & ~(size_t)255;

    __hip_bfloat16* Q = (__hip_bfloat16*)(ws);
    __hip_bfloat16* S = (__hip_bfloat16*)(ws + offS);
    unsigned long long* amax = (unsigned long long*)(ws + offA);

    init_kernel<<<(NB * NQ + 255) / 256, 256, 0, stream>>>(amax, out);

    for (int n = 0; n < NB; ++n) {
        int tq = MPAD * KDIM;
        pack_q_kernel<<<(tq + 255) / 256, 256, 0, stream>>>(ref_feat, Q, n);
        int ts = UPAD * KDIM;
        pack_s_kernel<<<(ts + 255) / 256, 256, 0, stream>>>(shape_feat, S, n);

        dim3 grid(MBLKS, UBLKS);
        gemm_argmax_kernel<<<grid, 256, SMEM_BYTES, stream>>>(Q, S, amax + (long)n * NQ);
    }

    loss_kernel<<<NB * NQ, 128, 0, stream>>>(output_feat, color_feat, amax, out);
}
